// CGU_22187801051687
// MI455X (gfx1250) — compile-verified, matrix operand order checked by reading, not run on hardware
//
#include <hip/hip_runtime.h>
#include <hip/hip_bf16.h>

// ---------------------------------------------------------------------------
// Model constants (from reference): B=16, N=64, T=256, D=64, DG=64, DS=9, NC=2
// F = D + 2*DG = 192, 2F = 384, HEADS = B*N = 1024.
// All heavy matmuls use v_wmma_f32_16x16x32_f16 (f16 in / f32 accumulate).
// Static operands (weights, V, graph-A) are pre-packed into WMMA fragment
// order so each B fragment is one contiguous 32B per-lane load (2x b128)
// instead of 16 scattered b16 loads. The 256-step GRU scan runs as ONE
// persistent kernel with a software grid barrier. Workspace: ~320 MB.
// ---------------------------------------------------------------------------
#define BB 16
#define NNODE 64
#define TT 256
#define DDIM 64
#define DGG 64
#define DSS 9
#define NCC 2
#define FFE 192
#define F2E 384
#define HEADS (BB * NNODE)
#define SCAN_BLOCKS 32  // 32 blocks * 8 waves = 256 waves, co-resident

typedef __attribute__((ext_vector_type(16))) _Float16 v16h;
typedef __attribute__((ext_vector_type(8)))  float    v8f;

// ----------------------- WMMA tile helpers (wave32) ------------------------
// A-matrix 16x32 f16 per-lane mapping (ISA 7.12.2): lane L: m=L&15, hi=L>>4;
// vgpr r (=e>>1): k = 16*(r>>2) + 8*hi + 2*(r&3) + (e&1).  Element pairs
// (2r, 2r+1) are contiguous (k, k+1) -> load pairwise so backend merges b32.
template <typename T>
__device__ __forceinline__ v16h load_Am(const T* src, int ld, int m0, int k0) {
  int lane = threadIdx.x & 31;
  int m = lane & 15, hi = lane >> 4;
  const T* row = src + (long)(m0 + m) * ld + k0;
  v16h a;
#pragma unroll
  for (int r = 0; r < 8; ++r) {
    int k = 16 * (r >> 2) + 8 * hi + 2 * (r & 3);
    a[2 * r]     = (_Float16)row[k];
    a[2 * r + 1] = (_Float16)row[k + 1];
  }
  return a;
}

// B (KxN'): symmetric striping, n = lane&15. BT=true reads B as src[n][k].
template <typename T, bool BT>
__device__ __forceinline__ v16h load_Bm(const T* src, int ld, int k0, int n0) {
  int lane = threadIdx.x & 31;
  int n = lane & 15, hi = lane >> 4;
  v16h b;
  if (BT) {
    const T* row = src + (long)(n0 + n) * ld + k0;
#pragma unroll
    for (int r = 0; r < 8; ++r) {
      int k = 16 * (r >> 2) + 8 * hi + 2 * (r & 3);
      b[2 * r]     = (_Float16)row[k];
      b[2 * r + 1] = (_Float16)row[k + 1];
    }
  } else {
#pragma unroll
    for (int r = 0; r < 8; ++r) {
      int k = 16 * (r >> 2) + 8 * hi + 2 * (r & 3);
      b[2 * r]     = (_Float16)src[(long)(k0 + k) * ld + n0 + n];
      b[2 * r + 1] = (_Float16)src[(long)(k0 + k + 1) * ld + n0 + n];
    }
  }
  return b;
}

// Fragment-order load: one 32B contiguous per-lane record per 32x16 tile.
__device__ __forceinline__ v16h load_frag(const _Float16* pk, long fragIdx) {
  return ((const v16h*)pk)[fragIdx * 32 + (threadIdx.x & 31)];
}

__device__ __forceinline__ v8f wmma16(const v16h& a, const v16h& b, const v8f& c) {
  return __builtin_amdgcn_wmma_f32_16x16x32_f16(false, a, false, b, (short)0, c,
                                                false, false);
}

// A gather + B gather (used where both operands are activations)
template <typename TA, typename TB, bool BT, int NT>
__device__ __forceinline__ void gemm_strip(const TA* A, int lda, int m0,
                                           const TB* Bp, int ldb, int n0,
                                           int K, v8f* acc) {
  for (int k0 = 0; k0 < K; k0 += 32) {
    if (k0 + 32 < K)
      __builtin_prefetch((const void*)(A + (long)(m0 + (threadIdx.x & 15)) * lda
                                       + k0 + 32), 0, 0);
    v16h a = load_Am<TA>(A, lda, m0, k0);
#pragma unroll
    for (int j = 0; j < NT; ++j) {
      v16h b = load_Bm<TB, BT>(Bp, ldb, k0, n0 + 16 * j);
      acc[j] = wmma16(a, b, acc[j]);
    }
  }
}

// A gather + B pre-packed fragments (weights): fragIdx = kc*ntCount + nt
template <typename TA, int NT>
__device__ __forceinline__ void gemm_strip_pk(const TA* A, int lda, int m0,
                                              const _Float16* Bpk, int ntCount,
                                              int nt0, int K, v8f* acc) {
  for (int k0 = 0, kc = 0; k0 < K; k0 += 32, ++kc) {
    if (k0 + 32 < K)
      __builtin_prefetch((const void*)(A + (long)(m0 + (threadIdx.x & 15)) * lda
                                       + k0 + 32), 0, 0);
    v16h a = load_Am<TA>(A, lda, m0, k0);
#pragma unroll
    for (int j = 0; j < NT; ++j)
      acc[j] = wmma16(a, load_frag(Bpk, (long)kc * ntCount + nt0 + j), acc[j]);
  }
}

// A pre-packed fragments + B gather (scan: A-matrix @ activations)
template <int NT>
__device__ __forceinline__ void gemm_strip_apk(const _Float16* Apk, int kcCount,
                                               int ms, const _Float16* Bsrc,
                                               int ldb, int n0, int K,
                                               v8f* acc) {
  for (int kc = 0; kc * 32 < K; ++kc) {
    v16h a = load_frag(Apk, (long)ms * kcCount + kc);
#pragma unroll
    for (int j = 0; j < NT; ++j) {
      v16h b = load_Bm<_Float16, false>(Bsrc, ldb, kc * 32, n0 + 16 * j);
      acc[j] = wmma16(a, b, acc[j]);
    }
  }
}

// ----------------------------- pack kernels --------------------------------
// Pack a KxN f32 weight into B-fragment order:
//   dst[((kc*(N/16) + nt)*32 + lane)*16 + e] = W[kc*32 + klocal][nt*16 + n]
__global__ void k_pack_frag(const float* W, _Float16* dst, int K, int N) {
  int i = blockIdx.x * 256 + threadIdx.x;
  if (i >= K * N) return;
  int e = i & 15, lane = (i >> 4) & 31, tile = i >> 9;
  int ntc = N >> 4;
  int nt = tile % ntc, kc = tile / ntc;
  int hi = lane >> 4, r = e >> 1, p = e & 1;
  int kl = 16 * (r >> 2) + 8 * hi + 2 * (r & 3) + p;
  int k = kc * 32 + kl, n = nt * 16 + (lane & 15);
  dst[i] = (_Float16)W[(long)k * N + n];
}

// Same, but for the fused [Wz|Wr] (K=384, N=128)
__global__ void k_pack_frag_zr(const float* Wz, const float* Wr, _Float16* dst) {
  int i = blockIdx.x * 256 + threadIdx.x;
  if (i >= F2E * 128) return;
  int e = i & 15, lane = (i >> 4) & 31, tile = i >> 9;
  int nt = tile & 7, kc = tile >> 3;
  int hi = lane >> 4, r = e >> 1, p = e & 1;
  int kl = 16 * (r >> 2) + 8 * hi + 2 * (r & 3) + p;
  int k = kc * 32 + kl, n = nt * 16 + (lane & 15);
  float v = (n < 64) ? Wz[(long)k * 64 + n] : Wr[(long)k * 64 + (n - 64)];
  dst[i] = (_Float16)v;
}

// Repack V (cols 128..191 of qkv) per head into B-fragment order:
//   frag = kc*4 + nt, kc over t-chunks (8), nt over d tiles (4)
__global__ void k_pack_v(const _Float16* qkv, _Float16* vpk) {
  int i = blockIdx.x * 256 + threadIdx.x;  // HEADS * 16384
  int head = i >> 14, rem = i & 16383;
  int e = rem & 15, lane = (rem >> 4) & 31, frag = rem >> 9;
  int nt = frag & 3, kc = frag >> 2;
  int hi = lane >> 4, r = e >> 1, p = e & 1;
  int kl = 16 * (r >> 2) + 8 * hi + 2 * (r & 3) + p;
  int t = kc * 32 + kl, d = nt * 16 + (lane & 15);
  vpk[i] = qkv[((long)head * TT + t) * 192 + 128 + d];
}

__global__ void k_zero_i(int* p, int n) {
  int i = threadIdx.x;
  if (i < n) p[i] = 0;
}

// ------------------------------ stage kernels ------------------------------
__global__ void k_build_pe(const float* x, float* pe) {
  int bt = blockIdx.x;  // b*T + t
  int d = threadIdx.x;  // 0..63
  float tm = x[(long)bt * 129 + 128];
  int j = d & 31;
  float ts = powf(256.0f, (float)j * (1.0f / 31.0f));
  float st = tm / ts;
  pe[(long)bt * 64 + d] = (d < 32) ? sinf(st) : cosf(st);
}

__global__ void k_build_tin(const float* x, const float* pw, const float* pb,
                            const float* pe, _Float16* tin) {
  int bt = blockIdx.x;
  int b = bt >> 8, t = bt & 255;
  int n = threadIdx.x >> 2;
  int d0 = (threadIdx.x & 3) * 16;
  float val = x[(long)bt * 129 + n];
  float obs = x[(long)bt * 129 + 64 + n];
  long base = ((long)(b * NNODE + n) * TT + t) * DDIM;
#pragma unroll
  for (int dd = 0; dd < 16; ++dd) {
    int d = d0 + dd;
    float v = val * pw[d] + obs * pw[64 + d] + pb[d] + pe[(long)bt * 64 + d];
    tin[base + d] = (_Float16)v;
  }
}

// qkv[head][t][0:192] = tin[head] @ Wqkv + bqkv         grid(HEADS, 8, 3)
__global__ void __launch_bounds__(64) k_qkv(const _Float16* tin,
                                            const _Float16* Wpk,
                                            const float* bias, _Float16* qkv) {
  int w = threadIdx.x >> 5, lane = threadIdx.x & 31;
  int head = blockIdx.x;
  int m0 = (blockIdx.y * 2 + w) * 16;
  int n0 = blockIdx.z * 64;
  v8f acc[4] = {};
  gemm_strip_pk<_Float16, 4>(tin + (long)head * TT * DDIM, DDIM, m0, Wpk, 12,
                             blockIdx.z * 4, DDIM, acc);
  _Float16* C = qkv + (long)head * TT * 3 * DDIM;
  int n = lane & 15, hi = lane >> 4;
#pragma unroll
  for (int j = 0; j < 4; ++j)
#pragma unroll
    for (int i = 0; i < 8; ++i) {
      int c = n0 + 16 * j + n;
      C[(long)(m0 + i + 8 * hi) * (3 * DDIM) + c] = (_Float16)(acc[j][i] + bias[c]);
    }
}

// attention per head: softmax(QK^T/8 + mask) @ V        grid(HEADS, 8)
__global__ void __launch_bounds__(64) k_attn(const _Float16* qkv,
                                             const _Float16* vpk,
                                             const int* lengths,
                                             _Float16* outO) {
  __shared__ float    Sls[2][16][256];
  __shared__ _Float16 Pls[2][16][256];
  int w = threadIdx.x >> 5, lane = threadIdx.x & 31;
  int head = blockIdx.x;
  int b = head >> 6;
  int m0 = (blockIdx.y * 2 + w) * 16;
  int len = lengths[b]; if (len < 1) len = 1;
  const _Float16* Q  = qkv + (long)head * TT * 192;
  const _Float16* Km = Q + 64;
  int n = lane & 15, hi = lane >> 4;

  // hoist Q fragments (K = 64 -> two 16x32 fragments)
  v16h qa0 = load_Am<_Float16>(Q, 192, m0, 0);
  v16h qa1 = load_Am<_Float16>(Q, 192, m0, 32);
  for (int n0 = 0; n0 < 256; n0 += 16) {
    v8f acc = {};
    acc = wmma16(qa0, load_Bm<_Float16, true>(Km, 192, 0, n0), acc);
    acc = wmma16(qa1, load_Bm<_Float16, true>(Km, 192, 32, n0), acc);
#pragma unroll
    for (int i = 0; i < 8; ++i) Sls[w][i + 8 * hi][n0 + n] = acc[i] * 0.125f;
  }
  __syncthreads();
  {
    int r = lane & 15;
    float mx = -1e30f;
    for (int c = hi * 128; c < hi * 128 + 128; ++c) {
      float v = (c < len) ? Sls[w][r][c] : -1e9f;
      mx = fmaxf(mx, v);
    }
    mx = fmaxf(mx, __shfl_xor(mx, 16, 32));
    float sum = 0.f;
    for (int c = hi * 128; c < hi * 128 + 128; ++c) {
      float v = (c < len) ? Sls[w][r][c] : -1e9f;
      sum += __expf(v - mx);
    }
    sum += __shfl_xor(sum, 16, 32);
    float inv = 1.0f / sum;
    for (int c = hi * 128; c < hi * 128 + 128; ++c) {
      float v = (c < len) ? Sls[w][r][c] : -1e9f;
      Pls[w][r][c] = (_Float16)(__expf(v - mx) * inv);
    }
  }
  __syncthreads();
  const _Float16* vp = vpk + (long)head * 16384;
  v8f acc[4] = {};
  for (int kc = 0; kc < 8; ++kc) {
    v16h a = load_Am<_Float16>(&Pls[w][0][0], 256, 0, kc * 32);
#pragma unroll
    for (int j = 0; j < 4; ++j)
      acc[j] = wmma16(a, load_frag(vp, kc * 4 + j), acc[j]);
  }
  _Float16* O = outO + (long)head * TT * DDIM;
#pragma unroll
  for (int j = 0; j < 4; ++j)
#pragma unroll
    for (int i = 0; i < 8; ++i)
      O[(long)(m0 + i + 8 * hi) * DDIM + j * 16 + n] = (_Float16)acc[j][i];
}

// C = resid + Ain@W + bias ; LayerNorm(C) ; PASS1 -> h1 ; PASS2 -> x_tat
template <int PASS>
__global__ void __launch_bounds__(64) k_gemm_ln(const _Float16* Ain,
                                                const _Float16* Wpk,
                                                const float* bias,
                                                const _Float16* resid,
                                                const float* g, const float* bn,
                                                _Float16* outp, const float* x,
                                                const _Float16* tin,
                                                _Float16* xtat) {
  __shared__ float Cls[2][16][64];
  int w = threadIdx.x >> 5, lane = threadIdx.x & 31;
  int head = blockIdx.x;
  int m0 = (blockIdx.y * 2 + w) * 16;
  v8f acc[4] = {};
  gemm_strip_pk<_Float16, 4>(Ain + (long)head * TT * DDIM, DDIM, m0, Wpk, 4, 0,
                             DDIM, acc);
  int n = lane & 15, hi = lane >> 4;
  const _Float16* R = resid + (long)head * TT * DDIM;
#pragma unroll
  for (int j = 0; j < 4; ++j)
#pragma unroll
    for (int i = 0; i < 8; ++i) {
      int m = m0 + i + 8 * hi, c = j * 16 + n;
      Cls[w][i + 8 * hi][c] = acc[j][i] + bias[c] + (float)R[(long)m * DDIM + c];
    }
  __syncthreads();
  int r = lane & 15;
  float s = 0.f, ss = 0.f;
  for (int c = hi * 32; c < hi * 32 + 32; ++c) {
    float v = Cls[w][r][c]; s += v; ss += v * v;
  }
  s  += __shfl_xor(s, 16, 32);
  ss += __shfl_xor(ss, 16, 32);
  float mean = s * (1.f / 64.f);
  float var  = ss * (1.f / 64.f) - mean * mean;
  float rstd = rsqrtf(var + 1e-5f);
  int m = m0 + r;  // time index t
  for (int c = hi * 32; c < hi * 32 + 32; ++c) {
    float hv = (Cls[w][r][c] - mean) * rstd * g[c] + bn[c];
    if (PASS == 1) {
      outp[((long)head * TT + m) * DDIM + c] = (_Float16)hv;
    } else {
      int bb = head >> 6, nn = head & 63;
      float obs = x[((long)bb * TT + m) * 129 + 64 + nn];
      float tv = (float)tin[((long)head * TT + m) * DDIM + c];
      xtat[((long)head * TT + m) * DDIM + c] =
          (_Float16)(obs * tv + (1.f - obs) * hv);
    }
  }
}

// ffmid = relu(h1 @ W1 + b1)                            grid(HEADS, 8)
__global__ void __launch_bounds__(64) k_ffmid(const _Float16* h1,
                                              const _Float16* Wpk,
                                              const float* bias,
                                              _Float16* outm) {
  int w = threadIdx.x >> 5, lane = threadIdx.x & 31;
  int head = blockIdx.x;
  int m0 = (blockIdx.y * 2 + w) * 16;
  v8f acc[4] = {};
  gemm_strip_pk<_Float16, 4>(h1 + (long)head * TT * DDIM, DDIM, m0, Wpk, 4, 0,
                             DDIM, acc);
  _Float16* C = outm + (long)head * TT * DDIM;
  int n = lane & 15, hi = lane >> 4;
#pragma unroll
  for (int j = 0; j < 4; ++j)
#pragma unroll
    for (int i = 0; i < 8; ++i) {
      int c = 16 * j + n;
      C[(long)(m0 + i + 8 * hi) * DDIM + c] =
          (_Float16)fmaxf(acc[j][i] + bias[c], 0.f);
    }
}

// xf[b][n][t] = nufft[b][1][t][n]
__global__ void k_xf(const float* nufft, float* xf) {
  int bt = blockIdx.x; int b = bt >> 8, t = bt & 255; int n = threadIdx.x;
  xf[((long)b * NNODE + n) * TT + t] = nufft[(((long)b * 2 + 1) * TT + t) * NNODE + n];
}

// Qf/Kf = xf[b] (64x256, f32) @ W (256x256) -> f16      grid(16, 2, 4)
__global__ void __launch_bounds__(64) k_proj_f(const float* xf,
                                               const _Float16* Wpk,
                                               _Float16* outp) {
  int w = threadIdx.x >> 5, lane = threadIdx.x & 31;
  int b = blockIdx.x;
  int m0 = (blockIdx.y * 2 + w) * 16;
  int n0 = blockIdx.z * 64;
  v8f acc[4] = {};
  gemm_strip_pk<float, 4>(xf + (long)b * NNODE * TT, TT, m0, Wpk, 16,
                          blockIdx.z * 4, TT, acc);
  _Float16* C = outp + (long)b * NNODE * TT;
  int n = lane & 15, hi = lane >> 4;
#pragma unroll
  for (int j = 0; j < 4; ++j)
#pragma unroll
    for (int i = 0; i < 8; ++i)
      C[(long)(m0 + i + 8 * hi) * TT + n0 + 16 * j + n] = (_Float16)acc[j][i];
}

// S[b] = relu(Qf @ Kf^T) / 16                           grid(16, 2)
__global__ void __launch_bounds__(64) k_S(const _Float16* Qf,
                                          const _Float16* Kf, float* S) {
  int w = threadIdx.x >> 5, lane = threadIdx.x & 31;
  int b = blockIdx.x;
  int m0 = (blockIdx.y * 2 + w) * 16;
  v8f acc[4] = {};
  gemm_strip<_Float16, _Float16, true, 4>(Qf + (long)b * NNODE * TT, TT, m0,
                                          Kf + (long)b * NNODE * TT, TT, 0, TT,
                                          acc);
  float* C = S + (long)b * NNODE * NNODE;
  int n = lane & 15, hi = lane >> 4;
#pragma unroll
  for (int j = 0; j < 4; ++j)
#pragma unroll
    for (int i = 0; i < 8; ++i)
      C[(long)(m0 + i + 8 * hi) * NNODE + 16 * j + n] =
          fmaxf(acc[j][i], 0.f) * 0.0625f;
}

// A = 0.5*(sm + sm^T), sm = softmax(0.5*(M+M^T)), M = mean_b softmax(S[b]).
// Emits A directly in WMMA A-fragment order: frag = (m>>4)*2 + (k>>5).
__global__ void k_A(const float* S, _Float16* A_pk) {
  __shared__ float A1[64][64], A2[64][64];
  int nrow = threadIdx.x;  // 64 threads
  float accrow[64];
#pragma unroll
  for (int m = 0; m < 64; ++m) accrow[m] = 0.f;
  for (int b = 0; b < BB; ++b) {
    const float* row = S + ((long)b * 64 + nrow) * 64;
    float mx = -1e30f;
    for (int m = 0; m < 64; ++m) mx = fmaxf(mx, row[m]);
    float sum = 0.f;
    for (int m = 0; m < 64; ++m) sum += __expf(row[m] - mx);
    float inv = 1.f / (sum * (float)BB);
    for (int m = 0; m < 64; ++m) accrow[m] += __expf(row[m] - mx) * inv;
  }
  for (int m = 0; m < 64; ++m) A1[nrow][m] = accrow[m];
  __syncthreads();
  for (int m = 0; m < 64; ++m) A2[nrow][m] = 0.5f * (A1[nrow][m] + A1[m][nrow]);
  __syncthreads();
  {
    float mx = -1e30f;
    for (int m = 0; m < 64; ++m) mx = fmaxf(mx, A2[nrow][m]);
    float sum = 0.f;
    for (int m = 0; m < 64; ++m) sum += __expf(A2[nrow][m] - mx);
    float inv = 1.f / sum;
    __syncthreads();
    for (int m = 0; m < 64; ++m) A1[nrow][m] = __expf(A2[nrow][m] - mx) * inv;
  }
  __syncthreads();
  int ms = nrow >> 4, ml = nrow & 15;
  for (int k = 0; k < 64; ++k) {
    float v = 0.5f * (A1[nrow][k] + A1[k][nrow]);
    int kc = k >> 5, kl = k & 31;
    int g = (kl >> 4) & 1, hi2 = (kl >> 3) & 1, rr = (kl >> 1) & 3, p = kl & 1;
    int lane2 = ml + 16 * hi2;
    int e = 2 * (g * 4 + rr) + p;
    A_pk[(((long)ms * 2 + kc) * 32 + lane2) * 16 + e] = (_Float16)v;
  }
}

// ----------------------- persistent GRU scan kernel ------------------------
__device__ __forceinline__ void grid_barrier(int* cnt, volatile int* gen,
                                             int nblk) {
  __threadfence();
  __syncthreads();
  if (threadIdx.x == 0) {
    int g = *gen;
    if (atomicAdd(cnt, 1) == nblk - 1) {
      *cnt = 0;
      __threadfence();
      atomicAdd((int*)gen, 1);
    } else {
      while (*gen == g) __builtin_amdgcn_s_sleep(1);
    }
    __threadfence();
  }
  __syncthreads();
}

__global__ void __launch_bounds__(256) k_scan(
    const _Float16* __restrict__ xtat, const float* __restrict__ pe,
    const float* __restrict__ delta_t, const _Float16* __restrict__ Apk,
    const _Float16* __restrict__ Wzrpk, const _Float16* __restrict__ Wcpk,
    const float* __restrict__ bz, const float* __restrict__ br,
    const float* __restrict__ bc, const int* __restrict__ lengths, float* h,
    _Float16* gin, _Float16* cin, float* zr, int* bar) {
  const int nblk = gridDim.x;
  const int tid = threadIdx.x;
  const int gidx = blockIdx.x * 256 + tid;
  const int stride = nblk * 256;
  const int gw = blockIdx.x * 8 + (tid >> 5);  // global wave id, 0..255
  const int lane = tid & 31;
  const int n = lane & 15, hi = lane >> 4;
  int* cnt = bar;
  volatile int* gen = bar + 1;

  // h = 0
  for (int i = gidx; i < HEADS * 64; i += stride) h[i] = 0.f;
  grid_barrier(cnt, gen, nblk);

  for (int t = 0; t < TT; ++t) {
    // P1: decay h, build gin[:, 0:192] = [x_t, h, te]
    for (int i = gidx; i < HEADS * 64; i += stride) {
      int bn = i >> 6, d = i & 63, b = bn >> 6;
      float decay = __expf(-fmaxf(delta_t[(long)b * TT + t], 0.f));
      float hv = h[i] * decay;
      h[i] = hv;
      _Float16* row = gin + (long)bn * F2E;
      row[d]       = xtat[((long)bn * TT + t) * DDIM + d];
      row[64 + d]  = (_Float16)hv;
      row[128 + d] = (_Float16)pe[((long)b * TT + t) * 64 + d];
    }
    grid_barrier(cnt, gen, nblk);

    // P2: gin[:, 192:384] = A @ gin[:, 0:192]  (192 wave tasks)
    if (gw < 192) {
      int b = gw / 12, rem = gw % 12;
      int m0 = (rem / 3) * 16;
      int n0 = (rem % 3) * 64;
      v8f acc[4] = {};
      gemm_strip_apk<4>(Apk, 2, m0 >> 4, gin + (long)b * NNODE * F2E, F2E, n0,
                        64, acc);
      _Float16* C = gin + (long)b * NNODE * F2E;
#pragma unroll
      for (int j = 0; j < 4; ++j)
#pragma unroll
        for (int i = 0; i < 8; ++i)
          C[(long)(m0 + i + 8 * hi) * F2E + 192 + n0 + 16 * j + n] =
              (_Float16)acc[j][i];
    }
    grid_barrier(cnt, gen, nblk);

    // P3: zr = sigmoid(gin @ [Wz|Wr] + [bz|br])  (128 wave tasks)
    if (gw < 128) {
      int m0 = (gw >> 1) * 16;
      int n0 = (gw & 1) * 64;
      v8f acc[4] = {};
      gemm_strip_pk<_Float16, 4>(gin, F2E, m0, Wzrpk, 8, (gw & 1) * 4, F2E,
                                 acc);
#pragma unroll
      for (int j = 0; j < 4; ++j)
#pragma unroll
        for (int i = 0; i < 8; ++i) {
          int c = n0 + 16 * j + n;
          float bias = (c < 64) ? bz[c] : br[c - 64];
          float v = acc[j][i] + bias;
          zr[(long)(m0 + i + 8 * hi) * 128 + c] = 1.f / (1.f + __expf(-v));
        }
    }
    grid_barrier(cnt, gen, nblk);

    // P4: cin[:, 0:192] = [x_t, r*h, te]
    for (int i = gidx; i < HEADS * 64; i += stride) {
      int bn = i >> 6, d = i & 63, b = bn >> 6;
      float r = zr[(long)bn * 128 + 64 + d];
      _Float16* row = cin + (long)bn * F2E;
      row[d]       = xtat[((long)bn * TT + t) * DDIM + d];
      row[64 + d]  = (_Float16)(r * h[i]);
      row[128 + d] = (_Float16)pe[((long)b * TT + t) * 64 + d];
    }
    grid_barrier(cnt, gen, nblk);

    // P5: cin[:, 192:384] = A @ cin[:, 0:192]
    if (gw < 192) {
      int b = gw / 12, rem = gw % 12;
      int m0 = (rem / 3) * 16;
      int n0 = (rem % 3) * 64;
      v8f acc[4] = {};
      gemm_strip_apk<4>(Apk, 2, m0 >> 4, cin + (long)b * NNODE * F2E, F2E, n0,
                        64, acc);
      _Float16* C = cin + (long)b * NNODE * F2E;
#pragma unroll
      for (int j = 0; j < 4; ++j)
#pragma unroll
        for (int i = 0; i < 8; ++i)
          C[(long)(m0 + i + 8 * hi) * F2E + 192 + n0 + 16 * j + n] =
              (_Float16)acc[j][i];
    }
    grid_barrier(cnt, gen, nblk);

    // P6: c = tanh(cin@Wc + bc); h = (t < len) ? z*h + (1-z)*c : h  (64 tasks)
    if (gw < 64) {
      int m0 = gw * 16;
      v8f acc[4] = {};
      gemm_strip_pk<_Float16, 4>(cin, F2E, m0, Wcpk, 4, 0, F2E, acc);
#pragma unroll
      for (int j = 0; j < 4; ++j)
#pragma unroll
        for (int i = 0; i < 8; ++i) {
          int m = m0 + i + 8 * hi;
          int c = 16 * j + n;
          float cv = tanhf(acc[j][i] + bc[c]);
          float z = zr[(long)m * 128 + c];
          float hv = h[(long)m * 64 + c];
          int b = m >> 6;
          int len = lengths[b]; if (len < 1) len = 1;
          float hn = z * hv + (1.f - z) * cv;
          h[(long)m * 64 + c] = (t < len) ? hn : hv;
        }
    }
    grid_barrier(cnt, gen, nblk);
  }
}

// ----------------------------- classifier head -----------------------------
__global__ void __launch_bounds__(256) k_final(const float* h, const float* Wfin,
                                               const float* bfin,
                                               const float* statc,
                                               const float* We, const float* be,
                                               const float* Wc1, const float* bc1,
                                               const float* Wc2, const float* bc2,
                                               float* out) {
  __shared__ float cat[BB][128];
  __shared__ float hcl[BB][200];
  int tid = threadIdx.x;
  for (int i = tid; i < HEADS; i += 256) {
    int b = i >> 6, nn = i & 63;
    float s = bfin[0];
    for (int d = 0; d < 64; ++d) s += h[(long)i * 64 + d] * Wfin[d];
    cat[b][nn] = s;
  }
  for (int i = tid; i < BB * 64; i += 256) {
    int b = i >> 6, j = i & 63;
    float s = be[j];
    for (int k = 0; k < DSS; ++k) s += statc[b * DSS + k] * We[k * 64 + j];
    cat[b][64 + j] = s;
  }
  __syncthreads();
  for (int i = tid; i < BB * 200; i += 256) {
    int b = i / 200, k = i % 200;
    float s = bc1[k];
    for (int f = 0; f < 128; ++f) s += cat[b][f] * Wc1[f * 200 + k];
    hcl[b][k] = fmaxf(s, 0.f);
  }
  __syncthreads();
  if (tid < BB * NCC) {
    int b = tid >> 1, c = tid & 1;
    float s = bc2[c];
    for (int k = 0; k < 200; ++k) s += hcl[b][k] * Wc2[k * 2 + c];
    out[b * 2 + c] = s;
  }
}

// ------------------------------- launcher ----------------------------------
extern "C" void kernel_launch(void* const* d_in, const int* in_sizes, int n_in,
                              void* d_out, int out_size, void* d_ws,
                              size_t ws_size, hipStream_t stream) {
  const float* x       = (const float*)d_in[0];
  const float* nufft   = (const float*)d_in[1];
  const float* delta_t = (const float*)d_in[2];
  const float* statc   = (const float*)d_in[3];
  const int*   lengths = (const int*)d_in[4];
  const float* proj_w  = (const float*)d_in[5];
  const float* proj_b  = (const float*)d_in[6];
  const float* Wqkv    = (const float*)d_in[7];
  const float* bqkv    = (const float*)d_in[8];
  const float* Wo      = (const float*)d_in[9];
  const float* bo      = (const float*)d_in[10];
  const float* ln1_g   = (const float*)d_in[11];
  const float* ln1_b   = (const float*)d_in[12];
  const float* W1      = (const float*)d_in[13];
  const float* b1      = (const float*)d_in[14];
  const float* W2      = (const float*)d_in[15];
  const float* b2      = (const float*)d_in[16];
  const float* ln2_g   = (const float*)d_in[17];
  const float* ln2_b   = (const float*)d_in[18];
  const float* WQ_f    = (const float*)d_in[19];
  const float* WK_f    = (const float*)d_in[20];
  const float* Wz      = (const float*)d_in[21];
  const float* bz      = (const float*)d_in[22];
  const float* Wr      = (const float*)d_in[23];
  const float* br      = (const float*)d_in[24];
  const float* Wc      = (const float*)d_in[25];
  const float* bc      = (const float*)d_in[26];
  const float* Wfin    = (const float*)d_in[27];
  const float* bfin    = (const float*)d_in[28];
  const float* We      = (const float*)d_in[29];
  const float* be      = (const float*)d_in[30];
  const float* Wc1     = (const float*)d_in[31];
  const float* bc1     = (const float*)d_in[32];
  const float* Wc2     = (const float*)d_in[33];
  const float* bc2     = (const float*)d_in[34];
  float* out = (float*)d_out;

  char* ws = (char*)d_ws;
  size_t off = 0;
  auto alloc = [&](size_t bytes) -> char* {
    char* p = ws + off;
    off = (off + bytes + 255) & ~(size_t)255;
    return p;
  };
  _Float16* tin_h   = (_Float16*)alloc((size_t)HEADS * TT * DDIM * 2);
  float*    pe      = (float*)   alloc((size_t)BB * TT * DDIM * 4);
  _Float16* qkv_h   = (_Float16*)alloc((size_t)HEADS * TT * 3 * DDIM * 2);
  _Float16* vpk_h   = (_Float16*)alloc((size_t)HEADS * TT * DDIM * 2);
  _Float16* attn_h  = (_Float16*)alloc((size_t)HEADS * TT * DDIM * 2);
  _Float16* h1_h    = (_Float16*)alloc((size_t)HEADS * TT * DDIM * 2);
  _Float16* ffmid_h = (_Float16*)alloc((size_t)HEADS * TT * DDIM * 2);
  _Float16* xtat_h  = (_Float16*)alloc((size_t)HEADS * TT * DDIM * 2);
  float*    xf      = (float*)   alloc((size_t)BB * NNODE * TT * 4);
  _Float16* qf_h    = (_Float16*)alloc((size_t)BB * NNODE * TT * 2);
  _Float16* kf_h    = (_Float16*)alloc((size_t)BB * NNODE * TT * 2);
  float*    Sbuf    = (float*)   alloc((size_t)BB * NNODE * NNODE * 4);
  _Float16* a_pk    = (_Float16*)alloc((size_t)NNODE * NNODE * 2);
  _Float16* wqkv_pk = (_Float16*)alloc((size_t)DDIM * 3 * DDIM * 2);
  _Float16* wo_pk   = (_Float16*)alloc((size_t)DDIM * DDIM * 2);
  _Float16* w1_pk   = (_Float16*)alloc((size_t)DDIM * DDIM * 2);
  _Float16* w2_pk   = (_Float16*)alloc((size_t)DDIM * DDIM * 2);
  _Float16* wqf_pk  = (_Float16*)alloc((size_t)TT * TT * 2);
  _Float16* wkf_pk  = (_Float16*)alloc((size_t)TT * TT * 2);
  _Float16* wzr_pk  = (_Float16*)alloc((size_t)F2E * 128 * 2);
  _Float16* wc_pk   = (_Float16*)alloc((size_t)F2E * 64 * 2);
  float*    hstate  = (float*)   alloc((size_t)HEADS * DGG * 4);
  _Float16* gin_h   = (_Float16*)alloc((size_t)HEADS * F2E * 2);
  _Float16* cin_h   = (_Float16*)alloc((size_t)HEADS * F2E * 2);
  float*    zr      = (float*)   alloc((size_t)HEADS * 128 * 4);
  int*      bar     = (int*)     alloc(2 * sizeof(int));

  // ---- weight packing into WMMA B-fragment order ----
  k_pack_frag<<<(12288 + 255) / 256, 256, 0, stream>>>(Wqkv, wqkv_pk, 64, 192);
  k_pack_frag<<<(4096 + 255) / 256, 256, 0, stream>>>(Wo, wo_pk, 64, 64);
  k_pack_frag<<<(4096 + 255) / 256, 256, 0, stream>>>(W1, w1_pk, 64, 64);
  k_pack_frag<<<(4096 + 255) / 256, 256, 0, stream>>>(W2, w2_pk, 64, 64);
  k_pack_frag<<<(65536 + 255) / 256, 256, 0, stream>>>(WQ_f, wqf_pk, 256, 256);
  k_pack_frag<<<(65536 + 255) / 256, 256, 0, stream>>>(WK_f, wkf_pk, 256, 256);
  k_pack_frag<<<(24576 + 255) / 256, 256, 0, stream>>>(Wc, wc_pk, 384, 64);
  k_pack_frag_zr<<<(49152 + 255) / 256, 256, 0, stream>>>(Wz, Wr, wzr_pk);
  k_zero_i<<<1, 32, 0, stream>>>(bar, 2);

  // ---- transformer over heads ----
  k_build_pe<<<BB * TT, 64, 0, stream>>>(x, pe);
  k_build_tin<<<BB * TT, 256, 0, stream>>>(x, proj_w, proj_b, pe, tin_h);
  k_qkv<<<dim3(HEADS, 8, 3), 64, 0, stream>>>(tin_h, wqkv_pk, bqkv, qkv_h);
  k_pack_v<<<(HEADS * TT * DDIM) / 256, 256, 0, stream>>>(qkv_h, vpk_h);
  k_attn<<<dim3(HEADS, 8), 64, 0, stream>>>(qkv_h, vpk_h, lengths, attn_h);
  k_gemm_ln<1><<<dim3(HEADS, 8), 64, 0, stream>>>(attn_h, wo_pk, bo, tin_h,
                                                  ln1_g, ln1_b, h1_h, x, tin_h,
                                                  xtat_h);
  k_ffmid<<<dim3(HEADS, 8), 64, 0, stream>>>(h1_h, w1_pk, b1, ffmid_h);
  k_gemm_ln<2><<<dim3(HEADS, 8), 64, 0, stream>>>(ffmid_h, w2_pk, b2, h1_h,
                                                  ln2_g, ln2_b, h1_h, x, tin_h,
                                                  xtat_h);

  // ---- graph adjacency A ----
  k_xf<<<BB * TT, 64, 0, stream>>>(nufft, xf);
  k_proj_f<<<dim3(BB, 2, 4), 64, 0, stream>>>(xf, wqf_pk, qf_h);
  k_proj_f<<<dim3(BB, 2, 4), 64, 0, stream>>>(xf, wkf_pk, kf_h);
  k_S<<<dim3(BB, 2), 64, 0, stream>>>(qf_h, kf_h, Sbuf);
  k_A<<<1, 64, 0, stream>>>(Sbuf, a_pk);

  // ---- gated graph-GRU scan: one persistent kernel, 256 steps ----
  k_scan<<<SCAN_BLOCKS, 256, 0, stream>>>(xtat_h, pe, delta_t, a_pk, wzr_pk,
                                          wc_pk, bz, br, bc, lengths, hstate,
                                          gin_h, cin_h, zr, bar);

  // ---- classifier head ----
  k_final<<<1, 256, 0, stream>>>(hstate, Wfin, bfin, statc, We, be, Wc1, bc1,
                                 Wc2, bc2, out);
}
